// StudentModel_37864431682201
// MI455X (gfx1250) — matrix operand order repaired and checked by reading, not compiled, not run on hardware
//
#include <hip/hip_runtime.h>
#include <hip/hip_bf16.h>

typedef __attribute__((ext_vector_type(16))) _Float16 v16h;
typedef __attribute__((ext_vector_type(8)))  _Float16 v8h;
typedef __attribute__((ext_vector_type(8)))  float    v8f;

#define BB  32          // batch
#define SS  512         // seq len
#define EE  512         // d_model
#define NHH 8           // heads
#define HD  64          // head dim
#define LL  3           // layers
#define MM  (BB*SS)     // 16384 rows of activations

// ---------------------------------------------------------------------------
// Fragment loaders for V_WMMA_F32_16X16X32_F16 (CDNA5 ISA 7.12.2, wave32).
// A: 16x32 (MxK). Lane l holds row m=l&15; halves 0..7 = K[k0+16g..+7],
//                 halves 8..15 = K[k0+16g+8..+7], g = l>>4.  (two 16B loads)
// B: 32x16 (KxN) stored as W[N][K] row-major: lane l holds col n=n0+(l&15);
//                 halves 0..15 = K[k0+16g .. k0+16g+15].     (one 32B load)
// C/D f32 16x16: lane l col n=l&15; VGPR r holds row m = r + 8*(l>>4).
// ---------------------------------------------------------------------------
__device__ __forceinline__ v16h load_a_frag(const _Float16* X, int ldx, int k0, int lane) {
    int m = lane & 15;
    int g = lane >> 4;
    const _Float16* p = X + (size_t)m * ldx + k0;
    v8h lo = *(const v8h*)(p + 16 * g);
    v8h hi = *(const v8h*)(p + 16 * g + 8);
    v16h r;
#pragma unroll
    for (int i = 0; i < 8; i++) { r[i] = lo[i]; r[8 + i] = hi[i]; }
    return r;
}

__device__ __forceinline__ v16h load_b_frag(const _Float16* W, int ldw, int n0, int k0, int lane) {
    int n = n0 + (lane & 15);
    int g = lane >> 4;
    return *(const v16h*)(W + (size_t)n * ldw + k0 + 16 * g);
}

// ---------------------------------------------------------------------------
// Generic WMMA GEMM: C[M,N] = A[M,K] * W[N,K]^T + bias (+ optional exact GELU)
// Optional f32 output Cf and/or f16 output Ch. Block = 8 waves; each wave
// computes a 16(M) x 64(N) tile (A fragment reused across 4 B fragments).
// The wave's whole 64xK weight strip is prefetched up-front (branchless
// global_prefetch_b8 loop) so the main K-loop stays a pure load/WMMA pipeline.
// ---------------------------------------------------------------------------
__global__ __launch_bounds__(256)
void wmma_gemm_kernel(const _Float16* __restrict__ A, const _Float16* __restrict__ W,
                      const float* __restrict__ bias, float* __restrict__ Cf,
                      _Float16* __restrict__ Ch, int M, int N, int K, int gelu)
{
    int lane = threadIdx.x & 31;
    int wave = threadIdx.x >> 5;
    int m0 = blockIdx.x * 16;
    int n0 = (blockIdx.y * 8 + wave) * 64;
    if (n0 >= N) return;                       // uniform per wave

    // up-front prefetch of this wave's weight strip (1 byte/lane per line set;
    // alternates 32-row halves so 4..16 instructions cover the 64xK strip)
    for (int pk = 0; pk < K * 2; pk += 256) {
        int prow = n0 + lane + ((pk >> 8) & 1) * 32;
        __builtin_prefetch((const void*)((const char*)(W + (size_t)prow * K) + pk), 0, 3);
    }

    const _Float16* Arow = A + (size_t)m0 * K;
    v8f acc[4] = {};

    for (int k0 = 0; k0 < K; k0 += 32) {
        v16h a = load_a_frag(Arow, K, k0, lane);
#pragma unroll
        for (int j = 0; j < 4; j++) {
            v16h b = load_b_frag(W, K, n0 + 16 * j, k0, lane);
            acc[j] = __builtin_amdgcn_wmma_f32_16x16x32_f16(
                false, a, false, b, (short)0, acc[j], false, false);
        }
    }

    int g = lane >> 4;
#pragma unroll
    for (int j = 0; j < 4; j++) {
        int n = n0 + 16 * j + (lane & 15);
        float bv = bias ? bias[n] : 0.0f;
#pragma unroll
        for (int r = 0; r < 8; r++) {
            int m = m0 + r + 8 * g;
            float v = acc[j][r] + bv;
            if (gelu) v = 0.5f * v * (1.0f + erff(v * 0.70710678118654752f));
            if (Cf) Cf[(size_t)m * N + n] = v;
            if (Ch) Ch[(size_t)m * N + n] = (_Float16)v;
        }
    }
}

// ---------------------------------------------------------------------------
// Per-head V transpose: vt[b,h,n,k] = V[b,k,h*64+n]  (f16).
// One 32(k) x 64(n) slab per block, transposed through LDS.
// Makes the attention A*V B-fragment a contiguous 32B load.
// ---------------------------------------------------------------------------
__global__ __launch_bounds__(256)
void vtrans_kernel(const _Float16* __restrict__ qkv, _Float16* __restrict__ vt)
{
    __shared__ __align__(16) _Float16 ts[32][66];
    int t  = threadIdx.x;
    int k0 = blockIdx.x * 32;
    int h  = blockIdx.y;
    int b  = blockIdx.z;
    const int rs = 3 * EE;
    const _Float16* Vb = qkv + (size_t)b * SS * rs + 2 * EE + h * HD;
    {
        int r = t >> 3;            // k row within slab: 0..31
        int c = (t & 7) * 8;       // n col: 0..56
        v8h v = *(const v8h*)(Vb + (size_t)(k0 + r) * rs + c);
        unsigned int* dst = (unsigned int*)&ts[r][c];        // 4B aligned
        const unsigned int* src = (const unsigned int*)&v;
#pragma unroll
        for (int i = 0; i < 4; i++) dst[i] = src[i];
    }
    __syncthreads();
    {
        int n  = t >> 2;           // 0..63
        int kk = (t & 3) * 8;      // 0..24
        v8h o;
#pragma unroll
        for (int i = 0; i < 8; i++) o[i] = ts[kk + i][n];
        *(v8h*)(vt + (((size_t)b * NHH + h) * HD + n) * SS + k0 + kk) = o;
    }
}

// ---------------------------------------------------------------------------
// Fused attention for one (batch, head, 16-query tile): 1 wave per block.
//   scores(16x512) = Q*K^T * 1/8 + key-padding bias   (WMMA, f32 in LDS)
//   softmax over keys (shfl_xor cross-half reduce)
//   probs -> d_out (f32, the huge [L,B,NH,S,S] output) + f16 copy in LDS
//   O(16x64) = probs * V  via pre-transposed V^T       (WMMA)
// ---------------------------------------------------------------------------
__global__ __launch_bounds__(32)
void attn_kernel(const _Float16* __restrict__ qkv, const _Float16* __restrict__ vt,
                 const int* __restrict__ mask,
                 float* __restrict__ attn_out, _Float16* __restrict__ o_h)
{
    __shared__ __align__(32) float    sc[16][SS];   // 32 KB scores / exp
    __shared__ __align__(32) _Float16 af[16][SS];   // 16 KB f16 probs
    __shared__ float sinv[16];

    int lane = threadIdx.x;
    int m0 = blockIdx.x * 16;
    int h  = blockIdx.y;
    int b  = blockIdx.z;
    int g  = lane >> 4;
    const int rs = 3 * EE;                                        // qkv row stride
    const _Float16* Qb  = qkv + ((size_t)b * SS + m0) * rs + h * HD;
    const _Float16* Kb  = qkv + (size_t)b * SS * rs + EE + h * HD;
    const _Float16* Vtb = vt + ((size_t)b * NHH + h) * HD * SS;   // [64][512]

    // ---- scores = Q K^T * scale + mask bias -> LDS ----
    for (int n0 = 0; n0 < SS; n0 += 64) {
        v8f acc[4] = {};
        for (int k0 = 0; k0 < HD; k0 += 32) {
            v16h a = load_a_frag(Qb, rs, k0, lane);
#pragma unroll
            for (int j = 0; j < 4; j++) {
                v16h bk = load_b_frag(Kb, rs, n0 + 16 * j, k0, lane);
                acc[j] = __builtin_amdgcn_wmma_f32_16x16x32_f16(
                    false, a, false, bk, (short)0, acc[j], false, false);
            }
        }
#pragma unroll
        for (int j = 0; j < 4; j++) {
            int n = n0 + 16 * j + (lane & 15);
            float mb = (mask[(size_t)b * SS + n] == 0) ? -1.0e30f : 0.0f;
#pragma unroll
            for (int r = 0; r < 8; r++)
                sc[r + 8 * g][n] = acc[j][r] * 0.125f + mb;
        }
    }
    __syncthreads();

    // ---- softmax: lane handles row=lane&15, half g of the 512 columns ----
    int row = lane & 15;
    int c0  = g * 256;
    float mx = -3.0e38f;
    for (int c = 0; c < 256; c++) mx = fmaxf(mx, sc[row][c0 + c]);
    mx = fmaxf(mx, __shfl_xor(mx, 16, 32));
    float sum = 0.0f;
    for (int c = 0; c < 256; c++) {
        float e = __expf(sc[row][c0 + c] - mx);
        sc[row][c0 + c] = e;
        sum += e;
    }
    sum += __shfl_xor(sum, 16, 32);
    if (lane < 16) sinv[lane] = 1.0f / sum;
    __syncthreads();

    // ---- write normalized probs (f32 -> d_out) and f16 copy -> LDS ----
    float* base = attn_out + (((size_t)b * NHH + h) * SS + m0) * SS;
    for (int idx = lane; idx < 16 * SS; idx += 32) {
        int r = idx >> 9, c = idx & (SS - 1);
        float v = sc[r][c] * sinv[r];
        base[(size_t)r * SS + c] = v;
        af[r][c] = (_Float16)v;
    }
    __syncthreads();

    // ---- O = probs * V (16 x 64), V^T rows are contiguous in k ----
    v8f oacc[4] = {};
    for (int k0 = 0; k0 < SS; k0 += 32) {
        v16h a = load_a_frag(&af[0][0], SS, k0, lane);
#pragma unroll
        for (int j = 0; j < 4; j++) {
            v16h bv = load_b_frag(Vtb, SS, 16 * j, k0, lane);
            oacc[j] = __builtin_amdgcn_wmma_f32_16x16x32_f16(
                false, a, false, bv, (short)0, oacc[j], false, false);
        }
    }
#pragma unroll
    for (int j = 0; j < 4; j++) {
        int n = 16 * j + (lane & 15);
#pragma unroll
        for (int r = 0; r < 8; r++) {
            int m = m0 + r + 8 * g;
            o_h[((size_t)b * SS + m) * EE + h * HD + n] = (_Float16)oacc[j][r];
        }
    }
}

// ---------------------------------------------------------------------------
// Embedding: x[b,s,:] = emb[ids[b,s],:] + pos[s,:]; writes f32 + f16 copies.
// ---------------------------------------------------------------------------
__global__ __launch_bounds__(256)
void embed_kernel(const int* __restrict__ ids, const float* __restrict__ emb,
                  const float* __restrict__ pos, float* __restrict__ x,
                  _Float16* __restrict__ xh)
{
    int row = blockIdx.x;            // b*S + s
    int s   = row & (SS - 1);
    int id  = ids[row];
    int t   = threadIdx.x;
#pragma unroll
    for (int i = 0; i < 2; i++) {
        int e = t + i * 256;
        float v = emb[(size_t)id * EE + e] + pos[(size_t)s * EE + e];
        x[(size_t)row * EE + e]  = v;
        xh[(size_t)row * EE + e] = (_Float16)v;
    }
}

// ---------------------------------------------------------------------------
// Residual + LayerNorm: x = LN(x + d) * g + b ; writes f32 (x) and f16 (xh).
// ---------------------------------------------------------------------------
__global__ __launch_bounds__(256)
void add_ln_kernel(float* __restrict__ x, const float* __restrict__ d,
                   const float* __restrict__ gam, const float* __restrict__ bet,
                   _Float16* __restrict__ xh)
{
    __shared__ float red[256];
    int row = blockIdx.x, t = threadIdx.x;
    float v0 = x[(size_t)row * EE + t]       + d[(size_t)row * EE + t];
    float v1 = x[(size_t)row * EE + t + 256] + d[(size_t)row * EE + t + 256];
    red[t] = v0 + v1;
    __syncthreads();
    for (int s2 = 128; s2 > 0; s2 >>= 1) { if (t < s2) red[t] += red[t + s2]; __syncthreads(); }
    float mean = red[0] * (1.0f / EE);
    __syncthreads();
    float d0 = v0 - mean, d1 = v1 - mean;
    red[t] = d0 * d0 + d1 * d1;
    __syncthreads();
    for (int s2 = 128; s2 > 0; s2 >>= 1) { if (t < s2) red[t] += red[t + s2]; __syncthreads(); }
    float rstd = rsqrtf(red[0] * (1.0f / EE) + 1e-5f);
    float y0 = d0 * rstd * gam[t]       + bet[t];
    float y1 = d1 * rstd * gam[t + 256] + bet[t + 256];
    x[(size_t)row * EE + t]        = y0;
    x[(size_t)row * EE + t + 256]  = y1;
    xh[(size_t)row * EE + t]       = (_Float16)y0;
    xh[(size_t)row * EE + t + 256] = (_Float16)y1;
}

// ---------------------------------------------------------------------------
// Classifier head: logits[b,c] = x[b,0,:] . cls_w[c,:] + cls_b[c]
// ---------------------------------------------------------------------------
__global__ __launch_bounds__(64)
void cls_kernel(const float* __restrict__ x, const float* __restrict__ w,
                const float* __restrict__ bb, float* __restrict__ out)
{
    int t = threadIdx.x;             // 0..63 -> (b, c)
    int b = t >> 1, c = t & 1;
    const float* xr = x + (size_t)b * SS * EE;     // CLS token s=0
    float s = bb[c];
    for (int e = 0; e < EE; e++) s += xr[e] * w[(size_t)c * EE + e];
    out[t] = s;
}

// f32 -> f16 weight conversion (once per launch; weights stay L2-resident)
__global__ __launch_bounds__(256)
void cvt_kernel(const float* __restrict__ s, _Float16* __restrict__ d, int n)
{
    int i = blockIdx.x * 256 + threadIdx.x;
    if (i < n) d[i] = (_Float16)s[i];
}

// ---------------------------------------------------------------------------
extern "C" void kernel_launch(void* const* d_in, const int* in_sizes, int n_in,
                              void* d_out, int out_size, void* d_ws, size_t ws_size,
                              hipStream_t stream)
{
    const int*   ids   = (const int*)d_in[0];
    const int*   amask = (const int*)d_in[1];
    const float* emb   = (const float*)d_in[2];
    const float* pos   = (const float*)d_in[3];
    const float* w_in  = (const float*)d_in[4];
    const float* b_in  = (const float*)d_in[5];
    const float* w_out = (const float*)d_in[6];
    const float* b_out = (const float*)d_in[7];
    const float* ln1g  = (const float*)d_in[8];
    const float* ln1b  = (const float*)d_in[9];
    const float* ln2g  = (const float*)d_in[10];
    const float* ln2b  = (const float*)d_in[11];
    const float* w_f1  = (const float*)d_in[12];
    const float* b_f1  = (const float*)d_in[13];
    const float* w_f2  = (const float*)d_in[14];
    const float* b_f2  = (const float*)d_in[15];
    const float* clsw  = (const float*)d_in[16];
    const float* clsb  = (const float*)d_in[17];

    char* ws = (char*)d_ws;
    size_t off = 0;
    auto take = [&](size_t bytes) -> char* {
        char* p = ws + off;
        off += (bytes + 255) & ~(size_t)255;
        return p;
    };
    _Float16* wqkv_h = (_Float16*)take((size_t)LL * 3 * EE * EE * 2);
    _Float16* wout_h = (_Float16*)take((size_t)LL * EE * EE * 2);
    _Float16* wf1_h  = (_Float16*)take((size_t)LL * 4 * EE * EE * 2);
    _Float16* wf2_h  = (_Float16*)take((size_t)LL * EE * 4 * EE * 2);
    float*    x      = (float*)   take((size_t)MM * EE * 4);
    _Float16* xh     = (_Float16*)take((size_t)MM * EE * 2);
    _Float16* qkvh   = (_Float16*)take((size_t)MM * 3 * EE * 2);
    _Float16* vt     = (_Float16*)take((size_t)BB * NHH * HD * SS * 2);
    _Float16* oh     = (_Float16*)take((size_t)MM * EE * 2);
    float*    tmp    = (float*)   take((size_t)MM * EE * 4);
    _Float16* fh     = (_Float16*)take((size_t)MM * 4 * EE * 2);

    float* logits = (float*)d_out;          // [32,2]
    float* attn   = logits + 64;            // [L,B,NH,S,S] f32

    auto cvt = [&](const float* s, _Float16* dptr, size_t n) {
        cvt_kernel<<<dim3((unsigned)((n + 255) / 256)), dim3(256), 0, stream>>>(s, dptr, (int)n);
    };
    cvt(w_in,  wqkv_h, (size_t)LL * 3 * EE * EE);
    cvt(w_out, wout_h, (size_t)LL * EE * EE);
    cvt(w_f1,  wf1_h,  (size_t)LL * 4 * EE * EE);
    cvt(w_f2,  wf2_h,  (size_t)LL * EE * 4 * EE);

    embed_kernel<<<dim3(MM), dim3(256), 0, stream>>>(ids, emb, pos, x, xh);

    for (int l = 0; l < LL; l++) {
        // QKV projection: [16384,512] x [1536,512]^T -> f16 qkv
        wmma_gemm_kernel<<<dim3(MM / 16, 3), dim3(256), 0, stream>>>(
            xh, wqkv_h + (size_t)l * 3 * EE * EE, b_in + (size_t)l * 3 * EE,
            nullptr, qkvh, MM, 3 * EE, EE, 0);

        // per-head V transpose (f16), feeds contiguous A*V fragments
        vtrans_kernel<<<dim3(SS / 32, NHH, BB), dim3(256), 0, stream>>>(qkvh, vt);

        // fused attention (softmax probs -> d_out, O -> f16)
        attn_kernel<<<dim3(SS / 16, NHH, BB), dim3(32), 0, stream>>>(
            qkvh, vt, amask, attn + (size_t)l * BB * NHH * SS * SS, oh);

        // out projection -> f32 for residual
        wmma_gemm_kernel<<<dim3(MM / 16, 1), dim3(256), 0, stream>>>(
            oh, wout_h + (size_t)l * EE * EE, b_out + (size_t)l * EE,
            tmp, nullptr, MM, EE, EE, 0);

        add_ln_kernel<<<dim3(MM), dim3(256), 0, stream>>>(
            x, tmp, ln1g + (size_t)l * EE, ln1b + (size_t)l * EE, xh);

        // FFN1 + exact GELU -> f16
        wmma_gemm_kernel<<<dim3(MM / 16, 4), dim3(256), 0, stream>>>(
            xh, wf1_h + (size_t)l * 4 * EE * EE, b_f1 + (size_t)l * 4 * EE,
            nullptr, fh, MM, 4 * EE, EE, 1);

        // FFN2 -> f32 for residual
        wmma_gemm_kernel<<<dim3(MM / 16, 1), dim3(256), 0, stream>>>(
            fh, wf2_h + (size_t)l * EE * 4 * EE, b_f2 + (size_t)l * EE,
            tmp, nullptr, MM, EE, 4 * EE, 0);

        add_ln_kernel<<<dim3(MM), dim3(256), 0, stream>>>(
            x, tmp, ln2g + (size_t)l * EE, ln2b + (size_t)l * EE, xh);
    }

    cls_kernel<<<dim3(1), dim3(64), 0, stream>>>(x, clsw, clsb, logits);
}
